// GraphMixer_Surv_66425964200432
// MI455X (gfx1250) — matrix operand order repaired and checked by reading, not compiled, not run on hardware
//
#include <hip/hip_runtime.h>
#include <hip/hip_bf16.h>
#include <math.h>

typedef __bf16 bf16;
typedef bf16  v16bf __attribute__((ext_vector_type(16)));
typedef float v8f   __attribute__((ext_vector_type(8)));

#define ACT_NONE 0
#define ACT_RELU 1
#define ACT_GELU 2
#define ACT_TANH 3
#define ACT_SIG  4

#if __has_builtin(__builtin_amdgcn_global_load_async_to_lds_b128) && \
    __has_builtin(__builtin_amdgcn_s_wait_asynccnt)
#define USE_ASYNC_LDS 1
#define AS1 __attribute__((address_space(1)))
#define AS3 __attribute__((address_space(3)))
typedef int v4i_a __attribute__((__vector_size__(16)));
#define ASYNC_CP16(g, l) __builtin_amdgcn_global_load_async_to_lds_b128( \
    (AS1 v4i_a*)(g), (AS3 v4i_a*)(l), 0, 0)
#else
#define USE_ASYNC_LDS 0
#endif

#if __has_builtin(__builtin_amdgcn_sched_group_barrier)
#define HAS_SGB 1
#else
#define HAS_SGB 0
#endif

__device__ __forceinline__ float apply_act(float x, int act) {
  switch (act) {
    case ACT_RELU: return x > 0.f ? x : 0.f;
    case ACT_GELU: return 0.5f * x * (1.f + erff(x * 0.70710678118654752f));
    case ACT_TANH: return tanhf(x);
    case ACT_SIG:  return 1.f / (1.f + expf(-x));
    default:       return x;
  }
}

__device__ __forceinline__ void atomicMaxFloat(float* addr, float val) {
  if (val >= 0.f) atomicMax((int*)addr, __float_as_int(val));
  else            atomicMin((unsigned int*)addr, (unsigned int)__float_as_int(val));
}

// ---------------------------------------------------------------------------
// Weight pre-swizzle: fp32 W[K,M] -> bf16 in WMMA B-fragment order:
//   [colGroup(M/128)][chunk(K/128)][ks(4)][ct(8)][lane(32)][elem(16)]
// ---------------------------------------------------------------------------
__global__ __launch_bounds__(256) void k_swz(
    const float* __restrict__ W, bf16* __restrict__ out, int K, int M)
{
  int idx = blockIdx.x * 256 + threadIdx.x;
  if (idx >= K * M) return;
  int k = idx / M, m = idx - k * M;
  int cg = m >> 7, mm = m & 127;
  int ch = k >> 7, kk = k & 127;
  int ks = kk >> 5, r = kk & 31;
  int ct = mm >> 4, c = mm & 15;
  int bl = (r & 8) ? (16 + c) : c;    // lanes 0-15: K 0..7,16..23 ; lanes 16-31: K 8..15,24..31
  size_t off = ((size_t)(cg * (K >> 7) + ch) * 16384)
             + (size_t)(((ks * 8 + ct) * 32 + bl) * 16 + ((r & 7) | ((r & 16) >> 1)));
  out[off] = (bf16)W[idx];
}

// ---------------------------------------------------------------------------
// Fused WMMA GEMM: C[N,M] = resid + act(A[N,K] @ B[K,M] + bias)
// Block = 256 threads (8 waves), block tile = 128x128, K chunked by 128.
// B pre-swizzled; LDS staging = contiguous copy, double-buffered (async when
// available). sched_group_barrier pins DS_READ/WMMA interleaving so each
// WMMA runs with the next fragment's loads in flight (staggered dscnt waits).
// ---------------------------------------------------------------------------
__global__ __launch_bounds__(256) void k_gemm_wmma(
    const float* __restrict__ A, const bf16* __restrict__ Bsw,
    const float* __restrict__ bias, const float* __restrict__ resid,
    float* __restrict__ C, int N, int K, int M, int act)
{
  __shared__ __align__(32) bf16 ldsB[2][16384];   // 2 x 32KB chunks

  const int tid  = threadIdx.x;
  const int lane = tid & 31;
  const int wave = tid >> 5;
  const int rowBase = blockIdx.x * 128 + wave * 16;
  const int colBase = blockIdx.y * 128;
  const int nChunks = K >> 7;
  const uint4* __restrict__ gB =
      (const uint4*)(Bsw + (size_t)blockIdx.y * nChunks * 16384);

  v8f acc[8];
  v8f zero = {};
  #pragma unroll
  for (int t = 0; t < 8; ++t) acc[t] = zero;

  // A-fragment (16-bit A 16x32 layout):
  // lane<16 : row=lane,    K 0..7 in elems 0-7, 16..23 in elems 8-15
  // lane>=16: row=lane-16, K 8..15 / 24..31
  const int ar = lane & 15;
  int rowA = rowBase + ar; if (rowA > N - 1) rowA = N - 1;  // clamp; EXEC stays full
  const float* __restrict__ pa = A + (size_t)rowA * K;
  const int kShift = (lane >> 4) << 3;

  auto buildA = [&](int k0) {
    v16bf af;
    #pragma unroll
    for (int i = 0; i < 8; ++i) af[i]     = (bf16)pa[k0 + i];
    #pragma unroll
    for (int i = 0; i < 8; ++i) af[8 + i] = (bf16)pa[k0 + 16 + i];
    return af;
  };
  auto ldB = [&](int buf, int ks, int ct) {
    return *(const v16bf*)&ldsB[buf][((ks * 8 + ct) * 32 + lane) * 16];
  };

  // ---- stage chunk 0 ----
#if USE_ASYNC_LDS
  {
    uint4* dst = (uint4*)ldsB[0];
    #pragma unroll
    for (int t = 0; t < 8; ++t)
      ASYNC_CP16(gB + t * 256 + tid, dst + t * 256 + tid);
  }
  __builtin_amdgcn_s_wait_asynccnt(0);
  __syncthreads();
#else
  {
    uint4* dst = (uint4*)ldsB[0];
    #pragma unroll
    for (int t = 0; t < 8; ++t) dst[t * 256 + tid] = gB[t * 256 + tid];
  }
  __syncthreads();
#endif

  for (int c = 0; c < nChunks; ++c) {
    const int buf = c & 1;
    const bool more = (c + 1 < nChunks);

#if USE_ASYNC_LDS
    if (more) {   // async-stage next chunk into the other buffer
      const uint4* src = gB + (size_t)(c + 1) * 2048;
      uint4* dst = (uint4*)ldsB[buf ^ 1];
      #pragma unroll
      for (int t = 0; t < 8; ++t)
        ASYNC_CP16(src + t * 256 + tid, dst + t * 256 + tid);
    }
#else
    uint4 stg[8];
    if (more) {   // register-stage next chunk
      const uint4* src = gB + (size_t)(c + 1) * 2048;
      #pragma unroll
      for (int t = 0; t < 8; ++t) stg[t] = src[t * 256 + tid];
    }
#endif

    const int kc = c << 7;
    #pragma unroll
    for (int ks = 0; ks < 4; ++ks) {
      v16bf af = buildA(kc + ks * 32 + kShift);
      v16bf bfr[8];
      #pragma unroll
      for (int ct = 0; ct < 8; ++ct) bfr[ct] = ldB(buf, ks, ct);
      #pragma unroll
      for (int ct = 0; ct < 8; ++ct)
        acc[ct] = __builtin_amdgcn_wmma_f32_16x16x32_bf16(
            false, af, false, bfr[ct], (short)0, acc[ct], false, false);
#if HAS_SGB
      // Pin the pipeline: L L | (L L W) x7 | W   (per 32-K step)
      // 0x100 = DS_READ ; 0x408 = MFMA|TRANS (WMMA classifies as either)
      __builtin_amdgcn_sched_group_barrier(0x100, 2, 0);
      #pragma unroll
      for (int i = 0; i < 7; ++i) {
        __builtin_amdgcn_sched_group_barrier(0x100, 2, 0);
        __builtin_amdgcn_sched_group_barrier(0x408, 1, 0);
      }
      __builtin_amdgcn_sched_group_barrier(0x408, 1, 0);
#endif
    }

    if (more) {
#if USE_ASYNC_LDS
      __builtin_amdgcn_s_wait_asynccnt(0);
      __syncthreads();
#else
      uint4* dst = (uint4*)ldsB[buf ^ 1];
      #pragma unroll
      for (int t = 0; t < 8; ++t) dst[t * 256 + tid] = stg[t];
      __syncthreads();
#endif
    }
  }

  // C/D layout: VGPR v -> row = (lane>=16 ? 8 : 0)+v, col = lane&15
  const int cc = lane & 15;
  const int rb = (lane >> 4) << 3;
  #pragma unroll
  for (int ct = 0; ct < 8; ++ct) {
    int col = colBase + ct * 16 + cc;
    #pragma unroll
    for (int v = 0; v < 8; ++v) {
      int row = rowBase + rb + v;
      if (row < N) {
        float x = acc[ct][v];
        if (bias)  x += bias[col];
        x = apply_act(x, act);
        if (resid) x += resid[(size_t)row * M + col];
        C[(size_t)row * M + col] = x;
      }
    }
  }
}

// ---------------------------------------------------------------------------
// LayerNorm over D=128, one wave per row
// ---------------------------------------------------------------------------
__global__ __launch_bounds__(256) void k_ln(
    const float* __restrict__ h, const float* __restrict__ g,
    const float* __restrict__ b, float* __restrict__ out, int N)
{
  int row = blockIdx.x * 8 + (threadIdx.x >> 5);
  if (row >= N) return;
  int lane = threadIdx.x & 31;
  float4 v = ((const float4*)(h + (size_t)row * 128))[lane];
  float s = v.x + v.y + v.z + v.w;
  #pragma unroll
  for (int off = 16; off > 0; off >>= 1) s += __shfl_xor(s, off, 32);
  float mean = s * (1.f / 128.f);
  float dx = v.x - mean, dy = v.y - mean, dz = v.z - mean, dw = v.w - mean;
  float q = dx * dx + dy * dy + dz * dz + dw * dw;
  #pragma unroll
  for (int off = 16; off > 0; off >>= 1) q += __shfl_xor(q, off, 32);
  float inv = rsqrtf(q * (1.f / 128.f) + 1e-5f);
  float4 gg = ((const float4*)g)[lane];
  float4 bb = ((const float4*)b)[lane];
  float4 o;
  o.x = dx * inv * gg.x + bb.x;
  o.y = dy * inv * gg.y + bb.y;
  o.z = dz * inv * gg.z + bb.z;
  o.w = dw * inv * gg.w + bb.w;
  ((float4*)(out + (size_t)row * 128))[lane] = o;
}

__global__ __launch_bounds__(256) void k_fill(float* __restrict__ p, float v, int n) {
  int i = blockIdx.x * 256 + threadIdx.x;
  if (i < n) p[i] = v;
}

// ---------------------------------------------------------------------------
// GATv2 edge kernels (E edges + N self loops)
// ---------------------------------------------------------------------------
__global__ __launch_bounds__(256) void k_edge_score(
    const int* __restrict__ ei, const float* __restrict__ xl,
    const float* __restrict__ xr, const float* __restrict__ att,
    float* __restrict__ ev, float* __restrict__ mx, int E, int N)
{
  int e = blockIdx.x * 8 + (threadIdx.x >> 5);
  if (e >= E + N) return;
  int lane = threadIdx.x & 31;
  int src = (e < E) ? ei[e]     : (e - E);
  int dst = (e < E) ? ei[E + e] : (e - E);
  float4 a = ((const float4*)(xl + (size_t)src * 128))[lane];
  float4 b = ((const float4*)(xr + (size_t)dst * 128))[lane];
  float4 t = ((const float4*)att)[lane];
  float sx = a.x + b.x; sx = sx > 0.f ? sx : 0.2f * sx;
  float sy = a.y + b.y; sy = sy > 0.f ? sy : 0.2f * sy;
  float sz = a.z + b.z; sz = sz > 0.f ? sz : 0.2f * sz;
  float sw = a.w + b.w; sw = sw > 0.f ? sw : 0.2f * sw;
  float s = sx * t.x + sy * t.y + sz * t.z + sw * t.w;
  #pragma unroll
  for (int off = 16; off > 0; off >>= 1) s += __shfl_xor(s, off, 32);
  if (lane == 0) {
    ev[e] = s;
    atomicMaxFloat(&mx[dst], s);
  }
}

__global__ __launch_bounds__(256) void k_edge_exp(
    const int* __restrict__ ei, float* __restrict__ ev,
    const float* __restrict__ mx, float* __restrict__ den, int E, int N)
{
  int e = blockIdx.x * 256 + threadIdx.x;
  if (e >= E + N) return;
  int dst = (e < E) ? ei[E + e] : (e - E);
  float v = expf(ev[e] - mx[dst]);
  ev[e] = v;
  atomicAdd(&den[dst], v);
}

__global__ __launch_bounds__(256) void k_edge_aggr(
    const int* __restrict__ ei, const float* __restrict__ ev,
    const float* __restrict__ den, const float* __restrict__ xl,
    float* __restrict__ agg, int E, int N)
{
  int e = blockIdx.x * 8 + (threadIdx.x >> 5);
  if (e >= E + N) return;
  int lane = threadIdx.x & 31;
  int src = (e < E) ? ei[e]     : (e - E);
  int dst = (e < E) ? ei[E + e] : (e - E);
  float alpha = ev[e] / (den[dst] + 1e-16f);
  float4 v = ((const float4*)(xl + (size_t)src * 128))[lane];
  float* po = agg + (size_t)dst * 128 + lane * 4;
  atomicAdd(po + 0, alpha * v.x);
  atomicAdd(po + 1, alpha * v.y);
  atomicAdd(po + 2, alpha * v.z);
  atomicAdd(po + 3, alpha * v.w);
}

__global__ __launch_bounds__(256) void k_gat_finish(
    float* __restrict__ h, const float* __restrict__ agg,
    const float* __restrict__ bias, int total)
{
  int i = blockIdx.x * 256 + threadIdx.x;
  if (i < total) h[i] += agg[i] + bias[i & 127];
}

// ---------------------------------------------------------------------------
// gated attention pooling
// ---------------------------------------------------------------------------
__global__ __launch_bounds__(256) void k_score(
    const float* __restrict__ ta, const float* __restrict__ sbuf,
    const float* __restrict__ cW, const float* __restrict__ cb,
    float* __restrict__ An, float* __restrict__ amax, int N)
{
  int n = blockIdx.x * 8 + (threadIdx.x >> 5);
  if (n >= N) return;
  int lane = threadIdx.x & 31;
  float4 a = ((const float4*)(ta   + (size_t)n * 128))[lane];
  float4 b = ((const float4*)(sbuf + (size_t)n * 128))[lane];
  float4 c = ((const float4*)cW)[lane];
  float s = a.x * b.x * c.x + a.y * b.y * c.y + a.z * b.z * c.z + a.w * b.w * c.w;
  #pragma unroll
  for (int off = 16; off > 0; off >>= 1) s += __shfl_xor(s, off, 32);
  if (lane == 0) {
    float val = s + cb[0];
    An[n] = val;
    atomicMaxFloat(amax, val);
  }
}

__global__ __launch_bounds__(256) void k_expsum(
    float* __restrict__ An, const float* __restrict__ amax,
    float* __restrict__ asum, int N)
{
  int n = blockIdx.x * 256 + threadIdx.x;
  if (n >= N) return;
  float v = expf(An[n] - *amax);
  An[n] = v;
  atomicAdd(asum, v);
}

__global__ __launch_bounds__(256) void k_pool(
    const float* __restrict__ An, const float* __restrict__ asum,
    const float* __restrict__ h, float* __restrict__ hp, int N)
{
  int n = blockIdx.x * 8 + (threadIdx.x >> 5);
  if (n >= N) return;
  int lane = threadIdx.x & 31;
  float w = An[n] / (*asum);
  float4 v = ((const float4*)(h + (size_t)n * 128))[lane];
  float* po = hp + lane * 4;
  atomicAdd(po + 0, w * v.x);
  atomicAdd(po + 1, w * v.y);
  atomicAdd(po + 2, w * v.z);
  atomicAdd(po + 3, w * v.w);
}

__global__ __launch_bounds__(128) void k_head(
    const float* __restrict__ hp, const float* __restrict__ rW,
    const float* __restrict__ rb, const float* __restrict__ cW,
    const float* __restrict__ cb, float* __restrict__ out)
{
  __shared__ float h2s[128];
  __shared__ float lg[4];
  int d = threadIdx.x;
  float acc = rb[d];
  #pragma unroll 4
  for (int k = 0; k < 128; ++k) acc += hp[k] * rW[k * 128 + d];
  h2s[d] = acc > 0.f ? acc : 0.f;
  __syncthreads();
  if (d < 4) {
    float s = cb[d];
    for (int k = 0; k < 128; ++k) s += h2s[k] * cW[k * 4 + d];
    lg[d] = s;
  }
  __syncthreads();
  if (d == 0) {
    float haz[4], S[4];
    float prod = 1.f; int best = 0; float bv = lg[0];
    for (int c = 0; c < 4; ++c) {
      haz[c] = 1.f / (1.f + expf(-lg[c]));
      prod *= (1.f - haz[c]);
      S[c] = prod;
      if (lg[c] > bv) { bv = lg[c]; best = c; }
    }
    for (int c = 0; c < 4; ++c) out[c]     = haz[c];
    for (int c = 0; c < 4; ++c) out[4 + c] = S[c];
    out[8] = (float)best;
    for (int c = 0; c < 4; ++c) out[9 + c] = lg[c];
  }
}

// ---------------------------------------------------------------------------
extern "C" void kernel_launch(void* const* d_in, const int* in_sizes, int n_in,
                              void* d_out, int out_size, void* d_ws, size_t ws_size,
                              hipStream_t stream)
{
  const float* x     = (const float*)d_in[0];
  const int*   ei    = (const int*)  d_in[1];
  const float* emb_W = (const float*)d_in[2];
  const float* emb_b = (const float*)d_in[3];
  const float* ln_g  = (const float*)d_in[4];
  const float* ln_b  = (const float*)d_in[5];
  const float* gWl   = (const float*)d_in[6];
  const float* gbl   = (const float*)d_in[7];
  const float* gWr   = (const float*)d_in[8];
  const float* gbr   = (const float*)d_in[9];
  const float* gatt  = (const float*)d_in[10];
  const float* gbias = (const float*)d_in[11];
  const float* W1a[3] = {(const float*)d_in[12], (const float*)d_in[14], (const float*)d_in[16]};
  const float* W2a[3] = {(const float*)d_in[13], (const float*)d_in[15], (const float*)d_in[17]};
  const float* aW = (const float*)d_in[18]; const float* ab = (const float*)d_in[19];
  const float* bW = (const float*)d_in[20]; const float* bb = (const float*)d_in[21];
  const float* cW = (const float*)d_in[22]; const float* cb = (const float*)d_in[23];
  const float* rW = (const float*)d_in[24]; const float* rb = (const float*)d_in[25];
  const float* clsW = (const float*)d_in[26]; const float* clsb = (const float*)d_in[27];
  float* out = (float*)d_out;

  const int N  = in_sizes[0] / 512;
  const int E  = in_sizes[1] / 2;
  const int EA = E + N;
  const int D  = 128;

  float* ws = (float*)d_ws;
  size_t o = 0;
  float* h    = ws + o; o += (size_t)N * D;
  float* hn   = ws + o; o += (size_t)N * D;
  float* xl   = ws + o; o += (size_t)N * D;
  float* xr   = ws + o; o += (size_t)N * D;
  float* agg  = ws + o; o += (size_t)N * D;
  float* ta   = ws + o; o += (size_t)N * D;
  float* sbuf = ws + o; o += (size_t)N * D;
  float* mid  = ws + o; o += (size_t)N * 3 * D;
  float* ev   = ws + o; o += (size_t)EA;
  float* mN   = ws + o; o += N;
  float* den  = ws + o; o += N;
  float* hp   = ws + o; o += D;
  float* amax = ws + o; o += 1;
  float* asum = ws + o; o += 1;
  o = (o + 3) & ~(size_t)3;          // 16B align for bf16 swizzled weights
  bf16* wsw = (bf16*)(ws + o);

  // swizzled weight offsets (bf16 elements)
  size_t so = 0;
  bf16* s_emb = wsw + so; so += (size_t)512 * 128;
  bf16* s_Wl[3]; for (int l = 0; l < 3; ++l) { s_Wl[l] = wsw + so; so += 128 * 128; }
  bf16* s_Wr[3]; for (int l = 0; l < 3; ++l) { s_Wr[l] = wsw + so; so += 128 * 128; }
  bf16* s_W1[3]; for (int l = 0; l < 3; ++l) { s_W1[l] = wsw + so; so += (size_t)128 * 128 * (l + 1); }
  bf16* s_W2[3]; for (int l = 0; l < 3; ++l) { s_W2[l] = wsw + so; so += (size_t)128 * 128 * (l + 1); }
  bf16* s_aW = wsw + so; so += 128 * 128;
  bf16* s_bW = wsw + so; so += 128 * 128;

  dim3 blk(256);
  const int gRows = (N + 127) / 128;
  auto swzGrid = [](int n) { return dim3((n + 255) / 256); };

  // ---- pre-swizzle all weights to bf16 fragment layout ----
  k_swz<<<swzGrid(512 * 128), blk, 0, stream>>>(emb_W, s_emb, 512, D);
  for (int l = 0; l < 3; ++l) {
    k_swz<<<swzGrid(128 * 128), blk, 0, stream>>>(gWl + (size_t)l * D * D, s_Wl[l], D, D);
    k_swz<<<swzGrid(128 * 128), blk, 0, stream>>>(gWr + (size_t)l * D * D, s_Wr[l], D, D);
    k_swz<<<swzGrid(128 * 128 * (l + 1)), blk, 0, stream>>>(W1a[l], s_W1[l], D, D * (l + 1));
    k_swz<<<swzGrid(128 * 128 * (l + 1)), blk, 0, stream>>>(W2a[l], s_W2[l], D * (l + 1), D);
  }
  k_swz<<<swzGrid(128 * 128), blk, 0, stream>>>(aW, s_aW, D, D);
  k_swz<<<swzGrid(128 * 128), blk, 0, stream>>>(bW, s_bW, D, D);

  // ---- h = relu(x @ emb_W + emb_b) ----
  k_gemm_wmma<<<dim3(gRows, 1), blk, 0, stream>>>(x, s_emb, emb_b, nullptr, h, N, 512, D, ACT_RELU);

  for (int l = 0; l < 3; ++l) {
    k_ln<<<dim3((N + 7) / 8), blk, 0, stream>>>(h, ln_g + l * D, ln_b + l * D, hn, N);
    k_gemm_wmma<<<dim3(gRows, 1), blk, 0, stream>>>(hn, s_Wl[l], gbl + l * D, nullptr, xl, N, D, D, ACT_NONE);
    k_gemm_wmma<<<dim3(gRows, 1), blk, 0, stream>>>(hn, s_Wr[l], gbr + l * D, nullptr, xr, N, D, D, ACT_NONE);
    k_fill<<<dim3((N + 255) / 256), blk, 0, stream>>>(mN, -INFINITY, N);
    k_fill<<<dim3((N + 255) / 256), blk, 0, stream>>>(den, 0.f, N);
    k_fill<<<dim3((N * D + 255) / 256), blk, 0, stream>>>(agg, 0.f, N * D);
    k_edge_score<<<dim3((EA + 7) / 8), blk, 0, stream>>>(ei, xl, xr, gatt + l * D, ev, mN, E, N);
    k_edge_exp<<<dim3((EA + 255) / 256), blk, 0, stream>>>(ei, ev, mN, den, E, N);
    k_edge_aggr<<<dim3((EA + 7) / 8), blk, 0, stream>>>(ei, ev, den, xl, agg, E, N);
    k_gat_finish<<<dim3((N * D + 255) / 256), blk, 0, stream>>>(h, agg, gbias + l * D, N * D);

    const int MM = D * (l + 1);
    k_gemm_wmma<<<dim3(gRows, MM / 128), blk, 0, stream>>>(h, s_W1[l], nullptr, nullptr,
                                                           mid, N, D, MM, ACT_GELU);
    k_gemm_wmma<<<dim3(gRows, 1), blk, 0, stream>>>(mid, s_W2[l], nullptr, h,
                                                    h, N, MM, D, ACT_NONE);
  }

  // ---- gated attention pooling ----
  k_gemm_wmma<<<dim3(gRows, 1), blk, 0, stream>>>(h, s_aW, ab, nullptr, ta, N, D, D, ACT_TANH);
  k_gemm_wmma<<<dim3(gRows, 1), blk, 0, stream>>>(h, s_bW, bb, nullptr, sbuf, N, D, D, ACT_SIG);
  k_fill<<<dim3(1), blk, 0, stream>>>(hp, 0.f, D);
  k_fill<<<dim3(1), blk, 0, stream>>>(amax, -INFINITY, 1);
  k_fill<<<dim3(1), blk, 0, stream>>>(asum, 0.f, 1);
  k_score<<<dim3((N + 7) / 8), blk, 0, stream>>>(ta, sbuf, cW, cb, ev, amax, N);
  k_expsum<<<dim3((N + 255) / 256), blk, 0, stream>>>(ev, amax, asum, N);
  k_pool<<<dim3((N + 7) / 8), blk, 0, stream>>>(ev, asum, h, hp, N);
  k_head<<<dim3(1), dim3(128), 0, stream>>>(hp, rW, rb, clsW, clsb, out);
}
